// CrossAttentionDebug_33724083208913
// MI455X (gfx1250) — matrix operand order repaired and checked by reading, not compiled, not run on hardware
//
#include <hip/hip_runtime.h>
#include <hip/hip_bf16.h>
#include <math.h>

typedef __attribute__((ext_vector_type(16))) _Float16 v16h;
typedef __attribute__((ext_vector_type(8)))  float    v8f;

#define WMMA_F32_F16(A_, B_, C_) \
  __builtin_amdgcn_wmma_f32_16x16x32_f16(false, (A_), false, (B_), (short)0, (C_), false, false)

static constexpr int EMBED = 1024;
static constexpr int HEADS = 16;
static constexpr int HDIM  = 64;
static constexpr int PROJ  = 64;   // phi feature dim = 2*PROJ = 128

// ---- CDNA5 async global->LDS copy (ASYNCcnt) with portable fallback --------
typedef int i32x4 __attribute__((ext_vector_type(4)));
typedef __attribute__((address_space(1))) i32x4 gi32x4;   // global
typedef __attribute__((address_space(3))) i32x4 li32x4;   // LDS

#if __has_builtin(__builtin_amdgcn_global_load_async_to_lds_b128)
#define HAVE_ASYNC_LDS 1
static __device__ __forceinline__ void async_cp16(void* lds, const void* g) {
  __builtin_amdgcn_global_load_async_to_lds_b128((gi32x4*)g, (li32x4*)lds, 0, 0);
}
#else
#define HAVE_ASYNC_LDS 0
static __device__ __forceinline__ void async_cp16(void* lds, const void* g) {
  *(float4*)lds = *(const float4*)g;   // synchronous fallback
}
#endif

#if __has_builtin(__builtin_amdgcn_s_wait_asynccnt)
#define WAIT_ASYNC(n) __builtin_amdgcn_s_wait_asynccnt(n)
#else
#define WAIT_ASYNC(n) asm volatile("s_wait_asynccnt %0" ::"i"(n) : "memory")
#endif

// ---------------------------------------------------------------------------
// C[M,N] = A[M,K] @ W[N,K]^T + bias[N]      (fp32 in/out, f16 WMMA inside)
// Block: 256 threads = 8 waves. Block tile 128(M) x 64(N); wave tile 16 x 64.
// W tile double-buffered in LDS via async global->LDS copies.
// ---------------------------------------------------------------------------
__global__ __launch_bounds__(256) void gemm_bias_kernel(
    const float* __restrict__ A, const float* __restrict__ W,
    const float* __restrict__ bias, float* __restrict__ C,
    int M, int N, int K)
{
  __shared__ float sB[2][64 * 32];          // 2 x (64 cols x 32 k) fp32 = 16 KB
  const int tid  = threadIdx.x;
  const int lane = tid & 31;
  const int wave = tid >> 5;
  const int rowBase = blockIdx.x * 128 + wave * 16;
  const int colBase = blockIdx.y * 64;
  const int m      = lane & 15;
  const int hi     = lane >> 4;             // 0 or 1 (lane half)
  const int kHalf  = hi << 3;               // 0 or 8  (A-fragment K base)
  const int kOff16 = hi << 4;               // 0 or 16 (unused)

  v8f acc[4] = {};
  const float* arow = A + (size_t)(rowBase + m) * K;

  // stage one 64x32 W tile into LDS buffer `buf` (2 x b128 per thread)
  auto stage = [&](int k0, int buf) {
    #pragma unroll
    for (int c = 0; c < 2; ++c) {
      const int li  = (tid + c * 256) * 4;  // float index within tile
      const int col = li >> 5;
      const int kk  = li & 31;
      async_cp16(&sB[buf][li], W + (size_t)(colBase + col) * K + k0 + kk);
    }
  };

  stage(0, 0);

  for (int k0 = 0; k0 < K; k0 += 32) {
    const int buf = (k0 >> 5) & 1;
    if (k0 + 32 < K) {
      stage(k0 + 32, buf ^ 1);   // prefetch next tile while current computes
      WAIT_ASYNC(2);             // oldest 2 (current tile) complete, next in flight
    } else {
      WAIT_ASYNC(0);
    }
    __syncthreads();

    // A fragment (16x32 f16): lanes 0-15 K={0..7,16..23}, lanes 16-31 K={8..15,24..31}
    v16h a;
    #pragma unroll
    for (int j = 0; j < 4; ++j) {
      a[2*j+0]   = (_Float16)arow[k0 + kHalf + 2*j + 0];
      a[2*j+1]   = (_Float16)arow[k0 + kHalf + 2*j + 1];
      a[8+2*j+0] = (_Float16)arow[k0 + 16 + kHalf + 2*j + 0];
      a[8+2*j+1] = (_Float16)arow[k0 + 16 + kHalf + 2*j + 1];
    }
    if (k0 + 32 < K) __builtin_prefetch(arow + k0 + 64, 0, 1);

    #pragma unroll
    for (int nt = 0; nt < 4; ++nt) {
      const float* bsrc = &sB[buf][(nt * 16 + m) * 32 + kHalf];
      v16h bf;
      #pragma unroll
      for (int i = 0; i < 16; ++i) bf[i] = (_Float16)bsrc[(i & 7) + ((i >> 3) << 4)];
      acc[nt] = WMMA_F32_F16(a, bf, acc[nt]);
    }
    __syncthreads();
  }

  #pragma unroll
  for (int nt = 0; nt < 4; ++nt) {
    const int col = colBase + nt * 16 + m;
    const float bv = bias[col];
    #pragma unroll
    for (int j = 0; j < 8; ++j) {
      const int row = rowBase + (hi << 3) + j;
      C[(size_t)row * N + col] = acc[nt][j] + bv;
    }
  }
}

// ---------------------------------------------------------------------------
// phi: X (R, H, 64) -> PHI (R, H, 128);  p = (X * D^-0.25) @ rm[h]^T,
// PHI = [sin(p), cos(p)] * PROJ^-0.5 ; rows with mask set are zeroed.
// Grid: (R/128, H); wave tile 16 rows x 64 proj.
// ---------------------------------------------------------------------------
__global__ __launch_bounds__(256) void phi_kernel(
    const float* __restrict__ X,
    const float* __restrict__ RM,                 // (H, 64, 64)
    const unsigned char* __restrict__ mask,       // (B, S) or nullptr
    float* __restrict__ PHI,
    int R, int Bsz)
{
  const int h    = blockIdx.y;
  const int tid  = threadIdx.x;
  const int lane = tid & 31;
  const int wave = tid >> 5;
  const int rowBase = blockIdx.x * 128 + wave * 16;
  const int m      = lane & 15;
  const int hi     = lane >> 4;
  const int kHalf  = hi << 3;
  const int kOff16 = hi << 4;
  const int S = R / Bsz;
  const float qs = 0.35355339059327373f;   // HEAD_DIM^-0.25
  const float ps = 0.125f;                 // PROJ^-0.5   (TAU == 1)

  v8f acc[4] = {};
  const float* xrow = X + ((size_t)(rowBase + m) * HEADS + h) * HDIM;

  #pragma unroll
  for (int k0 = 0; k0 < HDIM; k0 += 32) {
    v16h a;
    #pragma unroll
    for (int j = 0; j < 4; ++j) {
      a[2*j+0]   = (_Float16)(xrow[k0 + kHalf + 2*j + 0] * qs);
      a[2*j+1]   = (_Float16)(xrow[k0 + kHalf + 2*j + 1] * qs);
      a[8+2*j+0] = (_Float16)(xrow[k0 + 16 + kHalf + 2*j + 0] * qs);
      a[8+2*j+1] = (_Float16)(xrow[k0 + 16 + kHalf + 2*j + 1] * qs);
    }
    #pragma unroll
    for (int nt = 0; nt < 4; ++nt) {
      const float* wrow = RM + ((size_t)h * PROJ + nt * 16 + m) * HDIM + k0 + kHalf;
      v16h bf;
      #pragma unroll
      for (int i = 0; i < 16; ++i) bf[i] = (_Float16)wrow[(i & 7) + ((i >> 3) << 4)];
      acc[nt] = WMMA_F32_F16(a, bf, acc[nt]);
    }
  }

  #pragma unroll
  for (int nt = 0; nt < 4; ++nt) {
    const int kcol = nt * 16 + m;
    #pragma unroll
    for (int j = 0; j < 8; ++j) {
      const int row = rowBase + (hi << 3) + j;
      const float p = acc[nt][j];
      float sv = sinf(p) * ps;
      float cv = cosf(p) * ps;
      if (mask) {
        const int b = row % Bsz;
        const int s = row / Bsz;
        if (mask[(size_t)b * S + s]) { sv = 0.f; cv = 0.f; }
      }
      float* prow = PHI + ((size_t)row * HEADS + h) * 128;
      prow[kcol]      = sv;
      prow[64 + kcol] = cv;
    }
  }
}

// ---------------------------------------------------------------------------
// kv[b,h] (128 x 64) = phi_k[., b, h, :]^T @ v[., b, h, :]  over S keys
// ksum[b,h][128]     = sum_s phi_k
// One block per (b,h). Tiles staged in LDS transposed+converted to f16 so all
// fragment reads are contiguous ds loads; ksum reuses the staged tile.
// ---------------------------------------------------------------------------
__global__ __launch_bounds__(256) void kv_kernel(
    const float* __restrict__ PHIK,   // (S, B, H, 128)
    const float* __restrict__ V,      // (S, B, H, 64)
    float* __restrict__ KV,           // (B*H, 128, 64)
    float* __restrict__ KSUM,         // (B*H, 128)
    int S, int Bsz, int H)
{
  __shared__ _Float16 sPhiT[128][32];   // feature x s  (8 KB)
  __shared__ _Float16 sVT[64][32];      // d x s        (4 KB)
  const int bh = blockIdx.x;
  const int b  = bh / H;
  const int h  = bh % H;
  const int tid  = threadIdx.x;
  const int lane = tid & 31;
  const int wave = tid >> 5;
  const int m      = lane & 15;
  const int hi     = lane >> 4;
  const int kHalf  = hi << 3;
  const int kOff16 = hi << 4;
  const int mrow   = wave * 16 + m;       // feature index 0..127

  v8f acc[4] = {};
  float ksacc = 0.f;
  const int sRow = tid >> 3;              // 0..31
  const int f0   = (tid & 7) << 4;        // 0,16,..,112
  const int d0   = (tid & 7) << 3;        // 0,8,..,56

  for (int s0 = 0; s0 < S; s0 += 32) {
    // stage phi tile 32x128 -> sPhiT (transposed, f16)
    {
      const float* pr = PHIK + (((size_t)(s0 + sRow) * Bsz + b) * H + h) * 128 + f0;
      #pragma unroll
      for (int i = 0; i < 16; ++i) sPhiT[f0 + i][sRow] = (_Float16)pr[i];
      // stage v tile 32x64 -> sVT (transposed, f16)
      const float* vr = V + (((size_t)(s0 + sRow) * Bsz + b) * H + h) * HDIM + d0;
      #pragma unroll
      for (int i = 0; i < 8; ++i) sVT[d0 + i][sRow] = (_Float16)vr[i];
    }
    __syncthreads();

    // ksum partial from staged tile
    if (tid < 128) {
      #pragma unroll 8
      for (int s = 0; s < 32; ++s) ksacc += (float)sPhiT[tid][s];
    }

    // A fragment: A[feature][s]  (contiguous halves from sPhiT)
    v16h a;
    #pragma unroll
    for (int j = 0; j < 4; ++j) {
      a[2*j+0]   = sPhiT[mrow][kHalf + 2*j + 0];
      a[2*j+1]   = sPhiT[mrow][kHalf + 2*j + 1];
      a[8+2*j+0] = sPhiT[mrow][16 + kHalf + 2*j + 0];
      a[8+2*j+1] = sPhiT[mrow][16 + kHalf + 2*j + 1];
    }
    #pragma unroll
    for (int nt = 0; nt < 4; ++nt) {
      v16h bf;
      #pragma unroll
      for (int i = 0; i < 16; ++i) bf[i] = sVT[nt * 16 + m][kHalf + (i & 7) + ((i >> 3) << 4)];
      acc[nt] = WMMA_F32_F16(a, bf, acc[nt]);
    }
    __syncthreads();
  }

  #pragma unroll
  for (int nt = 0; nt < 4; ++nt) {
    #pragma unroll
    for (int j = 0; j < 8; ++j) {
      const int fr = wave * 16 + (hi << 3) + j;
      KV[((size_t)bh * 128 + fr) * HDIM + nt * 16 + m] = acc[nt][j];
    }
  }
  if (tid < 128) KSUM[(size_t)bh * 128 + tid] = ksacc;
}

// ---------------------------------------------------------------------------
// attn[t,b, h*64+d] = (phi_q[t,b,h,:] @ kv[b,h])[d] / (phi_q . ksum[b,h] + eps)
// Grid: (T/128, B*H). kv[b,h] staged once per block in LDS (transposed f16).
// ---------------------------------------------------------------------------
__global__ __launch_bounds__(256) void attn_kernel(
    const float* __restrict__ PHIQ,   // (T, B, H, 128)
    const float* __restrict__ KV,     // (B*H, 128, 64)
    const float* __restrict__ KSUM,   // (B*H, 128)
    float* __restrict__ OUT,          // (T, B, E)
    int T, int Bsz, int H)
{
  __shared__ _Float16 sKVT[64][128];    // d x feature (16 KB)
  __shared__ float sden[8][16];
  const int bh = blockIdx.y;
  const int b  = bh / H;
  const int h  = bh % H;
  const int tid  = threadIdx.x;
  const int lane = tid & 31;
  const int wave = tid >> 5;
  const int tBase = blockIdx.x * 128 + wave * 16;
  const int m      = lane & 15;
  const int hi     = lane >> 4;
  const int kHalf  = hi << 3;
  const int kOff16 = hi << 4;

  // stage kv[b,h] (128 x 64) -> sKVT transposed f16; 32 elements per thread
  {
    const int kRow = tid & 127;
    const int dBase = (tid >> 7) * 32;          // 0 or 32
    const float* src = KV + ((size_t)bh * 128 + kRow) * HDIM + dBase;
    #pragma unroll
    for (int i = 0; i < 32; ++i) sKVT[dBase + i][kRow] = (_Float16)src[i];
  }
  __syncthreads();

  v8f acc[4] = {};
  const float* qrow = PHIQ + (((size_t)(tBase + m) * Bsz + b) * H + h) * 128;

  #pragma unroll
  for (int k0 = 0; k0 < 128; k0 += 32) {
    v16h a;
    #pragma unroll
    for (int j = 0; j < 4; ++j) {
      a[2*j+0]   = (_Float16)qrow[k0 + kHalf + 2*j + 0];
      a[2*j+1]   = (_Float16)qrow[k0 + kHalf + 2*j + 1];
      a[8+2*j+0] = (_Float16)qrow[k0 + 16 + kHalf + 2*j + 0];
      a[8+2*j+1] = (_Float16)qrow[k0 + 16 + kHalf + 2*j + 1];
    }
    #pragma unroll
    for (int nt = 0; nt < 4; ++nt) {
      v16h bf;
      #pragma unroll
      for (int i = 0; i < 16; ++i) bf[i] = sKVT[nt * 16 + m][k0 + kHalf + (i & 7) + ((i >> 3) << 4)];
      acc[nt] = WMMA_F32_F16(a, bf, acc[nt]);
    }
  }

  // per-row denominator in fp32 (lanes 0..15, one row each)
  if (lane < 16) {
    const float* qr = PHIQ + (((size_t)(tBase + lane) * Bsz + b) * H + h) * 128;
    const float* ks = KSUM + (size_t)bh * 128;
    float d = 0.f;
    #pragma unroll 4
    for (int kk = 0; kk < 128; ++kk) d += qr[kk] * ks[kk];
    sden[wave][lane] = d;
  }
  __syncthreads();

  #pragma unroll
  for (int nt = 0; nt < 4; ++nt) {
    #pragma unroll
    for (int j = 0; j < 8; ++j) {
      const int r = (hi << 3) + j;
      const float denom = sden[wave][r] + 1e-6f;
      const int t = tBase + r;
      OUT[((size_t)t * Bsz + b) * EMBED + h * HDIM + nt * 16 + m] = acc[nt][j] / denom;
    }
  }
}

// ---------------------------------------------------------------------------
extern "C" void kernel_launch(void* const* d_in, const int* in_sizes, int n_in,
                              void* d_out, int out_size, void* d_ws, size_t ws_size,
                              hipStream_t stream)
{
  (void)in_sizes; (void)n_in; (void)out_size; (void)ws_size;

  const float* query = (const float*)d_in[0];   // (T, B, E)
  const float* key   = (const float*)d_in[1];   // (S, B, E)
  const float* rm    = (const float*)d_in[2];   // (H, 64, 64)
  const unsigned char* mask = (const unsigned char*)d_in[3]; // (B, S) bool
  const float* Wq = (const float*)d_in[4];
  const float* bq = (const float*)d_in[5];
  const float* Wk = (const float*)d_in[6];
  const float* bk = (const float*)d_in[7];
  const float* Wv = (const float*)d_in[8];
  const float* bv = (const float*)d_in[9];
  const float* Wo = (const float*)d_in[10];
  const float* bo = (const float*)d_in[11];
  float* out = (float*)d_out;

  const int T = 2048, S = 2048, B = 2;
  const int RQ = T * B;                    // 4096 rows
  const int RK = S * B;

  float* ws   = (float*)d_ws;
  float* qbuf = ws;                              // 4096*1024
  float* kbuf = qbuf + (size_t)RQ * EMBED;       // 4096*1024
  float* vbuf = kbuf + (size_t)RK * EMBED;       // 4096*1024
  float* phiq = vbuf + (size_t)RK * EMBED;       // 4096*16*128
  float* phik = phiq + (size_t)RQ * HEADS * 128; // 4096*16*128
  float* kvb  = phik + (size_t)RK * HEADS * 128; // 32*128*64
  float* ksum = kvb + (size_t)B * HEADS * 128 * HDIM; // 32*128
  float* attn = qbuf;                            // reuse q buffer (dead after phi_q)

  const dim3 blk(256);
  const dim3 gGemm(RQ / 128, EMBED / 64);        // (32, 16)
  const dim3 gPhi(RQ / 128, HEADS);              // (32, 16)
  const dim3 gKv(B * HEADS);                     // 32
  const dim3 gAttn(T / 128, B * HEADS);          // (16, 32)

  // Projections
  gemm_bias_kernel<<<gGemm, blk, 0, stream>>>(query, Wq, bq, qbuf, RQ, EMBED, EMBED);
  gemm_bias_kernel<<<gGemm, blk, 0, stream>>>(key,   Wk, bk, kbuf, RK, EMBED, EMBED);
  gemm_bias_kernel<<<gGemm, blk, 0, stream>>>(key,   Wv, bv, vbuf, RK, EMBED, EMBED);

  // Random-feature maps (mask folded into phi_k)
  phi_kernel<<<gPhi, blk, 0, stream>>>(qbuf, rm, nullptr, phiq, RQ, B);
  phi_kernel<<<gPhi, blk, 0, stream>>>(kbuf, rm, mask,    phik, RK, B);

  // Linear-attention key aggregation
  kv_kernel<<<gKv, blk, 0, stream>>>(phik, vbuf, kvb, ksum, S, B, HEADS);

  // Normalized attention output (T, B, E)
  attn_kernel<<<gAttn, blk, 0, stream>>>(phiq, kvb, ksum, attn, T, B, HEADS);

  // Output projection
  gemm_bias_kernel<<<gGemm, blk, 0, stream>>>(attn, Wo, bo, out, RQ, EMBED, EMBED);
}
